// GenericPyGModel_33191507263494
// MI455X (gfx1250) — matrix operand order repaired and checked by reading, not compiled, not run on hardware
//
#include <hip/hip_runtime.h>

#define N_NODES   100000
#define N_EDGES   1600000
#define F_INPUT   64
#define HID       128
#define NTILE     (HID / 16)     // 8 output tiles per 16-row strip
#define L_LAYERS  4
#define G_GRAPHS  512
#define FFN_DIM   300
#define BN_EPS    1e-5f

typedef __attribute__((ext_vector_type(16))) __bf16 v16bf;
typedef __attribute__((ext_vector_type(8)))  __bf16 v8bf;
typedef __attribute__((ext_vector_type(8)))  float  v8f;

__device__ __forceinline__ __bf16 f2bf(float f) {
  union { float f; unsigned u; } in; in.f = f;
  unsigned u = in.u;
  u += 0x7FFFu + ((u >> 16) & 1u);            // round-to-nearest-even
  unsigned short s = (unsigned short)(u >> 16);
  union { unsigned short s; __bf16 b; } out; out.s = s;
  return out.b;
}

// ---------------- degree / norm ----------------
__global__ void k_init_deg(float* deg) {
  int i = blockIdx.x * blockDim.x + threadIdx.x;
  if (i < N_NODES) deg[i] = 1.0f;             // self loop contributes 1
}
__global__ void k_deg_accum(const int* __restrict__ cols, float* deg) {
  int i = blockIdx.x * blockDim.x + threadIdx.x;
  if (i < N_EDGES) unsafeAtomicAdd(&deg[cols[i]], 1.0f);
}
__global__ void k_deg_rsqrt(float* deg) {
  int i = blockIdx.x * blockDim.x + threadIdx.x;
  if (i < N_NODES) deg[i] = rsqrtf(deg[i]);   // deg >= 1 always
}

// ---------------- utility ----------------
__global__ void k_zero_f32(float* p, int n) {
  int i = blockIdx.x * blockDim.x + threadIdx.x;
  if (i < n) p[i] = 0.0f;
}
__global__ void k_f32_to_bf16(const float* __restrict__ src, __bf16* __restrict__ dst, int n) {
  int i = blockIdx.x * blockDim.x + threadIdx.x;
  if (i < n) dst[i] = f2bf(src[i]);
}
// dst[n*K + k] = bf16(src[k*Nout + n])   (transpose + convert, for B operand)
__global__ void k_w_transpose_bf16(const float* __restrict__ src, __bf16* __restrict__ dst,
                                   int K, int Nout) {
  int i = blockIdx.x * blockDim.x + threadIdx.x;
  if (i >= K * Nout) return;
  int n = i / K, k = i % K;
  dst[i] = f2bf(src[(size_t)k * Nout + n]);
}

// ---------------- WMMA GEMM (strip form) ----------------
// C[M,HID] = A[M,K](bf16) @ Bt[HID,K](bf16)^T (+bias).
// One wave computes a 16-row x 128-col strip: A tile loaded ONCE per K-step,
// reused by 8 independent WMMA accumulators (no RAW chain between them).
// Optionally also emits a bf16 copy of C (feeds the next layer's A operand).
__global__ __launch_bounds__(256)
void k_gemm_strip(const __bf16* __restrict__ A, const __bf16* __restrict__ Bt,
                  const float* __restrict__ bias, float* __restrict__ C,
                  __bf16* __restrict__ Cbf, int M, int K) {
  int wave = blockIdx.x * (blockDim.x >> 5) + (threadIdx.x >> 5);
  if (wave >= (M >> 4)) return;               // uniform per wave -> EXEC all-1s inside
  int lane = threadIdx.x & 31;
  int half = lane >> 4;                       // 0: lanes 0-15, 1: lanes 16-31
  int mr   = lane & 15;

  const __bf16* arow = A + (size_t)(wave * 16 + mr) * K;   // A row m = lane&15

  v8f zero = {};
  v8f acc[NTILE];
#pragma unroll
  for (int t = 0; t < NTILE; ++t) acc[t] = zero;

  for (int k0 = 0; k0 < K; k0 += 32) {
    // 16-bit A layout (16x32): elems 0-15 -> K = 16h+0..15 (same order as B)
    v16bf a = *(const v16bf*)(arow + k0 + 16 * half);
#pragma unroll
    for (int t = 0; t < NTILE; ++t) {
      // B operand order used here (32x16, contiguous k, same as A): lane col n = lane&15; lanes 0-15 K=0..15, 16-31 K=16..31
      v16bf b = *(const v16bf*)(Bt + (size_t)(t * 16 + mr) * K + k0 + 16 * half);
      acc[t] = __builtin_amdgcn_wmma_f32_16x16x32_bf16(false, a, false, b, (short)0,
                                                       acc[t], false, false);
    }
  }

#pragma unroll
  for (int t = 0; t < NTILE; ++t) {
    float bv = bias ? bias[t * 16 + mr] : 0.0f;
    // C/D layout: lane half 0 -> M = v, half 1 -> M = v+8; N = lane&15
    size_t base = (size_t)(wave * 16 + 8 * half) * HID + t * 16 + mr;
#pragma unroll
    for (int v = 0; v < 8; ++v) {
      float val = acc[t][v] + bv;
      C[base + (size_t)v * HID] = val;
      if (Cbf) Cbf[base + (size_t)v * HID] = f2bf(val);
    }
  }
}

// ---------------- edge scatter (message passing) ----------------
// One wave per (edge or self-loop); each lane handles 4 of 128 features.
__global__ void k_scatter(const int* __restrict__ rows, const int* __restrict__ cols,
                          const float* __restrict__ dinv, const float* __restrict__ hw,
                          float* __restrict__ acc) {
  unsigned t = blockIdx.x * blockDim.x + threadIdx.x;
  int e = (int)(t >> 5);
  int lane = (int)(t & 31);
  const int total = N_EDGES + N_NODES;
  if (e >= total) return;
  int r, c; float nrm;
  if (e < N_EDGES) {
    r = rows[e]; c = cols[e];
    nrm = dinv[r] * dinv[c];
  } else {
    r = c = e - N_EDGES;
    float d = dinv[r];
    nrm = d * d;
  }
  float4 v = *(const float4*)(hw + (size_t)r * HID + lane * 4);
  float* dst = acc + (size_t)c * HID + lane * 4;
  unsafeAtomicAdd(dst + 0, nrm * v.x);
  unsafeAtomicAdd(dst + 1, nrm * v.y);
  unsafeAtomicAdd(dst + 2, nrm * v.z);
  unsafeAtomicAdd(dst + 3, nrm * v.w);
}

// ---- bias + BN(eval) + ReLU + optional residual; writes h (f32) and hbf (bf16) ----
__global__ void k_bn_relu_res(const float* __restrict__ acc, const float* __restrict__ bs,
                              const float* __restrict__ mean, const float* __restrict__ var,
                              const float* __restrict__ gamma, const float* __restrict__ beta,
                              float* __restrict__ h, __bf16* __restrict__ hbf, int residual) {
  int i = blockIdx.x * blockDim.x + threadIdx.x;
  if (i >= N_NODES * HID) return;
  int f = i & (HID - 1);
  float v = acc[i] + bs[f];
  v = (v - mean[f]) * rsqrtf(var[f] + BN_EPS) * gamma[f] + beta[f];
  v = fmaxf(v, 0.0f);
  if (residual) v += h[i];
  h[i]   = v;
  hbf[i] = f2bf(v);                 // free bf16 copy for next layer's GEMM A-operand
}

// ---------------- global mean pool ----------------
__global__ void k_pool_count(const int* __restrict__ batch, float* cnts) {
  int i = blockIdx.x * blockDim.x + threadIdx.x;
  if (i < N_NODES) unsafeAtomicAdd(&cnts[batch[i]], 1.0f);
}
__global__ void k_pool_sum(const float* __restrict__ h, const int* __restrict__ batch,
                           float* sums) {
  int i = blockIdx.x * blockDim.x + threadIdx.x;
  if (i >= N_NODES * HID) return;
  int n = i >> 7, f = i & (HID - 1);
  unsafeAtomicAdd(&sums[(size_t)batch[n] * HID + f], h[i]);
}
__global__ void k_pool_div(float* sums, const float* __restrict__ cnts) {
  int i = blockIdx.x * blockDim.x + threadIdx.x;
  if (i >= G_GRAPHS * HID) return;
  sums[i] = sums[i] / fmaxf(cnts[i >> 7], 1.0f);
}

// ---------------- small dense layers (MLP head) ----------------
__global__ void k_dense(const float* __restrict__ in, const float* __restrict__ W,
                        const float* __restrict__ b, float* __restrict__ out,
                        int Gn, int Kin, int Nout, int doRelu) {
  int idx = blockIdx.x * blockDim.x + threadIdx.x;
  if (idx >= Gn * Nout) return;
  int g = idx / Nout, j = idx - g * Nout;
  float s = b[j];
  const float* r = in + (size_t)g * Kin;
  for (int k = 0; k < Kin; ++k) s += r[k] * W[(size_t)k * Nout + j];
  if (doRelu) s = fmaxf(s, 0.0f);
  out[idx] = s;
}

extern "C" void kernel_launch(void* const* d_in, const int* in_sizes, int n_in,
                              void* d_out, int out_size, void* d_ws, size_t ws_size,
                              hipStream_t stream) {
  const float* x     = (const float*)d_in[0];
  const int*   ei    = (const int*)d_in[1];
  const int*   batch = (const int*)d_in[2];
  const float* encW  = (const float*)d_in[3];
  const float* encB  = (const float*)d_in[4];
  const float* Ws    = (const float*)d_in[5];
  const float* bs    = (const float*)d_in[6];
  const float* gam   = (const float*)d_in[7];
  const float* bet   = (const float*)d_in[8];
  const float* mean  = (const float*)d_in[9];
  const float* var   = (const float*)d_in[10];
  const float* W0    = (const float*)d_in[11];
  const float* b0    = (const float*)d_in[12];
  const float* W1    = (const float*)d_in[13];
  const float* b1    = (const float*)d_in[14];
  const float* W2    = (const float*)d_in[15];
  const float* b2    = (const float*)d_in[16];
  float* out = (float*)d_out;

  const int* rows = ei;
  const int* cols = ei + N_EDGES;

  // ---- carve workspace ----
  char* ws = (char*)d_ws;
  size_t off = 0;
  auto take = [&](size_t bytes) -> void* {
    void* p = ws + off;
    off = (off + bytes + 255) & ~(size_t)255;
    return p;
  };
  const int NH = N_NODES * HID;
  float*  dinv  = (float*) take((size_t)N_NODES * 4);
  float*  h     = (float*) take((size_t)NH * 4);
  float*  hw    = (float*) take((size_t)NH * 4);
  float*  acc   = (float*) take((size_t)NH * 4);
  __bf16* hbf   = (__bf16*)take((size_t)NH * 2);
  __bf16* xbf   = (__bf16*)take((size_t)N_NODES * F_INPUT * 2);
  __bf16* encWt = (__bf16*)take((size_t)HID * F_INPUT * 2);
  __bf16* Wt    = (__bf16*)take((size_t)L_LAYERS * HID * HID * 2);
  float*  sums  = (float*) take((size_t)G_GRAPHS * HID * 4);
  float*  cnts  = (float*) take((size_t)G_GRAPHS * 4);
  float*  t0    = (float*) take((size_t)G_GRAPHS * FFN_DIM * 4);
  float*  t1    = (float*) take((size_t)G_GRAPHS * FFN_DIM * 4);
  (void)ws_size; (void)in_sizes; (void)n_in; (void)out_size;

  const int B = 256;
  auto blk = [](long long n, int b) { return (int)((n + b - 1) / b); };
  const int gemmBlocks = blk(N_NODES / 16, 8);   // 8 waves (strips) per 256-thread block

  // ---- GCN normalization ----
  k_init_deg  <<<blk(N_NODES, B), B, 0, stream>>>(dinv);
  k_deg_accum <<<blk(N_EDGES, B), B, 0, stream>>>(cols, dinv);
  k_deg_rsqrt <<<blk(N_NODES, B), B, 0, stream>>>(dinv);

  // ---- prep bf16 operands ----
  k_f32_to_bf16     <<<blk((long long)N_NODES * F_INPUT, B), B, 0, stream>>>(x, xbf, N_NODES * F_INPUT);
  k_w_transpose_bf16<<<blk(HID * F_INPUT, B), B, 0, stream>>>(encW, encWt, F_INPUT, HID);
  for (int i = 0; i < L_LAYERS; ++i)
    k_w_transpose_bf16<<<blk(HID * HID, B), B, 0, stream>>>(
        Ws + (size_t)i * HID * HID, Wt + (size_t)i * HID * HID, HID, HID);

  // ---- encoder: h = x @ encW + encB (also emits bf16 copy for layer-0 GEMM) ----
  k_gemm_strip<<<gemmBlocks, B, 0, stream>>>(xbf, encWt, encB, h, hbf, N_NODES, F_INPUT);

  // ---- GCN layers ----
  const long long scatterThreads = (long long)(N_EDGES + N_NODES) * 32;
  for (int i = 0; i < L_LAYERS; ++i) {
    k_gemm_strip<<<gemmBlocks, B, 0, stream>>>(
        hbf, Wt + (size_t)i * HID * HID, nullptr, hw, nullptr, N_NODES, HID);
    k_zero_f32<<<blk(NH, B), B, 0, stream>>>(acc, NH);
    k_scatter <<<blk(scatterThreads, B), B, 0, stream>>>(rows, cols, dinv, hw, acc);
    k_bn_relu_res<<<blk(NH, B), B, 0, stream>>>(
        acc, bs + i * HID, mean + i * HID, var + i * HID,
        gam + i * HID, bet + i * HID, h, hbf, i > 0 ? 1 : 0);
  }

  // ---- global mean pool ----
  k_zero_f32 <<<blk(G_GRAPHS * HID, B), B, 0, stream>>>(sums, G_GRAPHS * HID);
  k_zero_f32 <<<blk(G_GRAPHS, B), B, 0, stream>>>(cnts, G_GRAPHS);
  k_pool_count<<<blk(N_NODES, B), B, 0, stream>>>(batch, cnts);
  k_pool_sum  <<<blk(NH, B), B, 0, stream>>>(h, batch, sums);
  k_pool_div  <<<blk(G_GRAPHS * HID, B), B, 0, stream>>>(sums, cnts);

  // ---- MLP head ----
  k_dense<<<blk(G_GRAPHS * FFN_DIM, B), B, 0, stream>>>(sums, W0, b0, t0,
                                                        G_GRAPHS, HID, FFN_DIM, 1);
  k_dense<<<blk(G_GRAPHS * FFN_DIM, B), B, 0, stream>>>(t0, W1, b1, t1,
                                                        G_GRAPHS, FFN_DIM, FFN_DIM, 1);
  k_dense<<<blk(G_GRAPHS, B), B, 0, stream>>>(t1, W2, b2, out,
                                              G_GRAPHS, FFN_DIM, 1, 0);
}
